// Attention_86328842649726
// MI455X (gfx1250) — matrix operand order repaired and checked by reading, not compiled, not run on hardware
//
#include <hip/hip_runtime.h>

// ---------- types ----------
typedef __bf16 bf16;
typedef __attribute__((ext_vector_type(16))) __bf16 v16bf;
typedef __attribute__((ext_vector_type(8)))  float  v8f;
typedef __attribute__((ext_vector_type(4)))  unsigned int u32x4;

#define DEV __device__ __forceinline__

// MLA config (matches reference)
#define CFG_B   2
#define CFG_S   2048
#define CFG_D   1024
#define CFG_H   16
#define CFG_DH  64
#define CFG_L   256
#define NTOK    (CFG_B * CFG_S)   // 4096

DEV bf16 f2bf(float f) {
  unsigned u = __builtin_bit_cast(unsigned, f);
  u += 0x7FFFu + ((u >> 16) & 1u);          // round-to-nearest-even
  unsigned short h = (unsigned short)(u >> 16);
  return __builtin_bit_cast(bf16, h);
}
DEV unsigned short bf_bits(float f) {
  return __builtin_bit_cast(unsigned short, f2bf(f));
}

DEV v8f vzero8() {
  v8f z;
#pragma unroll
  for (int e = 0; e < 8; ++e) z[e] = 0.0f;
  return z;
}

// ---------- WMMA fragment loaders (contiguous k = 16*(lane/16) + s, wave32) ----------
// A: 16x32 bf16 (MxK), source row-major [M][K] with leading dim ld
DEV v16bf load_frag_a(const bf16* __restrict__ p, int ld) {
  const int lane = threadIdx.x & 31;
  const int row  = lane & 15;
  const int koff = (lane >> 4) << 4;        // 0 or 16
  v16bf f;
#pragma unroll
  for (int i = 0; i < 8; ++i) f[i]     = p[row * ld + koff + i];
#pragma unroll
  for (int i = 0; i < 8; ++i) f[8 + i] = p[row * ld + 8 + koff + i];
  return f;
}

// B: 32x16 bf16 (KxN) where the source is stored N-major ([N][K], ld = lead):
DEV v16bf load_frag_bT(const bf16* __restrict__ p, int ld) {
  const int lane = threadIdx.x & 31;
  const int n    = lane & 15;
  const int koff = (lane >> 4) << 4;        // 0 or 16
  v16bf f;
#pragma unroll
  for (int i = 0; i < 16; ++i) f[i] = p[n * ld + koff + i];
  return f;
}

DEV v8f wmma_bf16(v16bf a, v16bf b, v8f c) {
  return __builtin_amdgcn_wmma_f32_16x16x32_bf16(
      false, a, false, b, (short)0, c, false, false);
}

// ---------- fp32 -> bf16 conversion ----------
__global__ void f32_to_bf16_kernel(const float* __restrict__ in,
                                   bf16* __restrict__ out, int n) {
  int i = blockIdx.x * blockDim.x + threadIdx.x;
  if (i < n) out[i] = f2bf(in[i]);
}

// fp32 [R][C] -> bf16 transposed [C][R] (one-time weight preparation)
__global__ void f32_to_bf16_T_kernel(const float* __restrict__ in,
                                     bf16* __restrict__ out, int R, int C) {
  int i = blockIdx.x * blockDim.x + threadIdx.x;
  if (i < R * C) {
    int r = i / C, c = i % C;
    out[(size_t)c * R + r] = f2bf(in[i]);
  }
}

// ---------- tiled WMMA GEMM: C[MxN] = A[MxK] @ Bt^T + bias ----------
// Bt is the transposed B, layout [N][K]. BM=128, BN=64, BK=32; 128 threads =
// 4 waves stacked in M, each wave computes 32x64 (2x4 fragments, 8 WMMA/step).
// Software-pipelined: global loads for tile k+1 issued before tile k's WMMAs.
// MODE: 0 = bf16 row-major, 1 = f32 row-major, 2 = bf16 V-transposed
template<int MODE>
__global__ __launch_bounds__(128) void gemm_bf16_kernel(
    const bf16* __restrict__ A, const bf16* __restrict__ Bt,
    const float* __restrict__ bias,
    bf16* __restrict__ Cb, float* __restrict__ Cf,
    int M, int N, int K)
{
  constexpr int BM = 128, BN = 64, BK = 32;
  constexpr int LDT = BK + 8;               // 40 bf16 = 80B rows (16B-aligned)
  __shared__ __align__(16) bf16 sA[BM * LDT];
  __shared__ __align__(16) bf16 sBt[BN * LDT];

  const int tiles_n = N / BN;
  const int tm = (int)blockIdx.x / tiles_n;
  const int tn = (int)blockIdx.x % tiles_n;
  const int tid  = threadIdx.x;
  const int lane = tid & 31;
  const int wave = tid >> 5;
  const int wm   = wave * 32;

  // fixed per-thread staging assignment: chunk (cc) -> row ar+cc*32, col ac
  const int ar = tid >> 2;                  // 0..31
  const int ac = (tid & 3) * 8;             // 0,8,16,24
  const bf16* pA[4];
  const bf16* pB[2];
#pragma unroll
  for (int cc = 0; cc < 4; ++cc)
    pA[cc] = A + (size_t)(tm * BM + ar + cc * 32) * K + ac;
#pragma unroll
  for (int cc = 0; cc < 2; ++cc)
    pB[cc] = Bt + (size_t)(tn * BN + ar + cc * 32) * K + ac;

  u32x4 ra[4], rb[2];
#pragma unroll
  for (int cc = 0; cc < 4; ++cc) ra[cc] = *(const u32x4*)(pA[cc]);
#pragma unroll
  for (int cc = 0; cc < 2; ++cc) rb[cc] = *(const u32x4*)(pB[cc]);

  v8f acc[2][4];
#pragma unroll
  for (int i = 0; i < 2; ++i)
#pragma unroll
    for (int j = 0; j < 4; ++j) acc[i][j] = vzero8();

  for (int k0 = 0; k0 < K; k0 += BK) {
    // commit staged registers to LDS
#pragma unroll
    for (int cc = 0; cc < 4; ++cc)
      *(u32x4*)(sA + (ar + cc * 32) * LDT + ac) = ra[cc];
#pragma unroll
    for (int cc = 0; cc < 2; ++cc)
      *(u32x4*)(sBt + (ar + cc * 32) * LDT + ac) = rb[cc];
    __syncthreads();

    // issue next tile's global loads (overlap with WMMAs below)
    if (k0 + BK < K) {
#pragma unroll
      for (int cc = 0; cc < 4; ++cc)
        ra[cc] = *(const u32x4*)(pA[cc] + k0 + BK);
#pragma unroll
      for (int cc = 0; cc < 2; ++cc)
        rb[cc] = *(const u32x4*)(pB[cc] + k0 + BK);
    }

    v16bf a0 = load_frag_a(sA + (wm + 0)  * LDT, LDT);
    v16bf a1 = load_frag_a(sA + (wm + 16) * LDT, LDT);
    v16bf bfr[4];
#pragma unroll
    for (int j = 0; j < 4; ++j)
      bfr[j] = load_frag_bT(sBt + (j * 16) * LDT, LDT);
#pragma unroll
    for (int j = 0; j < 4; ++j) {
      acc[0][j] = wmma_bf16(a0, bfr[j], acc[0][j]);
      acc[1][j] = wmma_bf16(a1, bfr[j], acc[1][j]);
    }
    __syncthreads();
  }

#pragma unroll
  for (int i = 0; i < 2; ++i)
#pragma unroll
    for (int j = 0; j < 4; ++j) {
      int gcol = tn * BN + j * 16 + (lane & 15);
      float bv = bias ? bias[gcol] : 0.0f;
#pragma unroll
      for (int e = 0; e < 8; ++e) {
        int grow = tm * BM + wm + i * 16 + e + ((lane >> 4) << 3);
        float v = acc[i][j][e] + bv;
        if (MODE == 0) {
          Cb[(size_t)grow * N + gcol] = f2bf(v);
        } else if (MODE == 1) {
          Cf[(size_t)grow * N + gcol] = v;
        } else {
          int b = grow >> 11, s = grow & (CFG_S - 1);
          int h = gcol >> 6, d = gcol & 63;
          Cb[(((size_t)(b * CFG_H + h) * CFG_DH + d) << 11) + s] = f2bf(v);
        }
      }
    }
}

// ---------- flash attention (non-causal, per (b, h, 64-row q tile)) ----------
// Q/K: [NTOK, H*DH] bf16 row-major.  V: [B][H][DH][S] bf16 (transposed).
// 128 threads = 4 waves; wave w owns rows [w*16, w*16+16) of the q tile.
__global__ __launch_bounds__(128) void mla_flash_kernel(
    const bf16* __restrict__ Q, const bf16* __restrict__ Kt,
    const bf16* __restrict__ Vt, bf16* __restrict__ O)
{
  constexpr int S = CFG_S, H = CFG_H, Dh = CFG_DH;
  constexpr int Br = 64, Bc = 64, NTILE = S / Bc;
  constexpr int LQ = Dh + 8;                // 72 bf16 = 144B rows
  constexpr int LP = Bc + 8;                // 72
  constexpr float SCALE = 0.125f;           // 64^-0.5

  __shared__ __align__(16) bf16 sQ[Br * LQ];
  __shared__ __align__(16) bf16 sK[Bc * LQ];     // [key][Dh]
  __shared__ __align__(16) bf16 sVt[Dh * LP];    // [d][key]
  __shared__ __align__(16) float sS[4][16 * Bc];
  __shared__ __align__(16) bf16 sP[4][16 * LP];
  __shared__ __align__(16) float sM[4][16], sL[4][16], sAl[4][16];

  const int tid = threadIdx.x, lane = tid & 31, wave = tid >> 5;
  const int half = lane >> 4;               // 0 or 1
  const int qt = (int)blockIdx.x % (S / Br);
  const int h  = ((int)blockIdx.x / (S / Br)) % H;
  const int b  = (int)blockIdx.x / ((S / Br) * H);
  const size_t rs    = (size_t)H * Dh;                   // 1024
  const size_t base  = (size_t)b * S * rs + (size_t)h * Dh;
  const size_t vbase = (size_t)(b * H + h) * Dh * S;

  // fixed per-thread staging assignment: row sr+cc*16, col sc8
  const int sr  = tid >> 3;                 // 0..15
  const int sc8 = (tid & 7) * 8;            // 0..56
  const bf16* kp[4];
  const bf16* vp[4];
#pragma unroll
  for (int cc = 0; cc < 4; ++cc) {
    kp[cc] = Kt + base + (size_t)(sr + cc * 16) * rs + sc8;
    vp[cc] = Vt + vbase + (size_t)(sr + cc * 16) * S + sc8;
  }

  // stage Q tile once
#pragma unroll
  for (int cc = 0; cc < 4; ++cc)
    *(u32x4*)(sQ + (sr + cc * 16) * LQ + sc8) =
        *(const u32x4*)(Q + base + (size_t)(qt * Br + sr + cc * 16) * rs + sc8);
  if (lane < 16) { sM[wave][lane] = -1e30f; sL[wave][lane] = 0.0f; }

  // preload first K/V tile into registers
  u32x4 kreg[4], vreg[4];
#pragma unroll
  for (int cc = 0; cc < 4; ++cc) {
    kreg[cc] = *(const u32x4*)(kp[cc]);
    vreg[cc] = *(const u32x4*)(vp[cc]);
  }

  v8f o[4];
#pragma unroll
  for (int nf = 0; nf < 4; ++nf) o[nf] = vzero8();
  __syncthreads();

  for (int t = 0; t < NTILE; ++t) {
    // commit staged K/V registers to LDS
#pragma unroll
    for (int cc = 0; cc < 4; ++cc) {
      *(u32x4*)(sK  + (sr + cc * 16) * LQ + sc8) = kreg[cc];
      *(u32x4*)(sVt + (sr + cc * 16) * LP + sc8) = vreg[cc];
    }
    __syncthreads();

    // issue next tile's global loads (overlap with compute)
    if (t + 1 < NTILE) {
#pragma unroll
      for (int cc = 0; cc < 4; ++cc) {
        kreg[cc] = *(const u32x4*)(kp[cc] + (size_t)(t + 1) * Bc * rs);
        vreg[cc] = *(const u32x4*)(vp[cc] + (size_t)(t + 1) * Bc);
      }
    }

    // scores: (16 x Bc) = Qstrip (16 x Dh) @ K^T
    v8f sc[4];
#pragma unroll
    for (int nf = 0; nf < 4; ++nf) sc[nf] = vzero8();
#pragma unroll
    for (int ks = 0; ks < Dh; ks += 32) {
      v16bf aq = load_frag_a(sQ + (wave * 16) * LQ + ks, LQ);
#pragma unroll
      for (int nf = 0; nf < 4; ++nf) {
        v16bf bk = load_frag_bT(sK + (nf * 16) * LQ + ks, LQ);
        sc[nf] = wmma_bf16(aq, bk, sc[nf]);
      }
    }
#pragma unroll
    for (int nf = 0; nf < 4; ++nf)
#pragma unroll
      for (int e = 0; e < 8; ++e) {
        int r = e + (half << 3);
        sS[wave][r * Bc + nf * 16 + (lane & 15)] = sc[nf][e] * SCALE;
      }
    __syncthreads();

    // online softmax: each half-wave handles 32 of the 64 columns of row
    // r = lane&15; halves combined with shfl_xor(16).
    {
      const int r = lane & 15;
      const float4* rp = (const float4*)&sS[wave][r * Bc + half * 32];
      float mo = sM[wave][r];
      float mx = -1e30f;
#pragma unroll
      for (int c = 0; c < 8; ++c) {
        float4 v4 = rp[c];
        mx = fmaxf(mx, fmaxf(fmaxf(v4.x, v4.y), fmaxf(v4.z, v4.w)));
      }
      mx = fmaxf(mx, __shfl_xor(mx, 16));
      mx = fmaxf(mx, mo);
      float sum = 0.0f;
#pragma unroll
      for (int c0 = 0; c0 < 32; c0 += 8) {
        float4 u = rp[c0 / 4], w = rp[c0 / 4 + 1];
        float p0 = __expf(u.x - mx), p1 = __expf(u.y - mx);
        float p2 = __expf(u.z - mx), p3 = __expf(u.w - mx);
        float p4 = __expf(w.x - mx), p5 = __expf(w.y - mx);
        float p6 = __expf(w.z - mx), p7 = __expf(w.w - mx);
        sum += (p0 + p1) + (p2 + p3) + (p4 + p5) + (p6 + p7);
        u32x4 pk;
        pk[0] = (unsigned)bf_bits(p0) | ((unsigned)bf_bits(p1) << 16);
        pk[1] = (unsigned)bf_bits(p2) | ((unsigned)bf_bits(p3) << 16);
        pk[2] = (unsigned)bf_bits(p4) | ((unsigned)bf_bits(p5) << 16);
        pk[3] = (unsigned)bf_bits(p6) | ((unsigned)bf_bits(p7) << 16);
        *(u32x4*)&sP[wave][r * LP + half * 32 + c0] = pk;
      }
      sum += __shfl_xor(sum, 16);
      if (half == 0) {
        float al = __expf(mo - mx);
        sM[wave][r]  = mx;
        sL[wave][r]  = sL[wave][r] * al + sum;
        sAl[wave][r] = al;
      }
    }
    __syncthreads();

    // rescale accumulators (vectorized alpha fetch: 8 rows of this half)
    {
      const float4* ap = (const float4*)&sAl[wave][half * 8];
      float4 a0 = ap[0], a1 = ap[1];
      float alv[8] = {a0.x, a0.y, a0.z, a0.w, a1.x, a1.y, a1.z, a1.w};
#pragma unroll
      for (int nf = 0; nf < 4; ++nf)
#pragma unroll
        for (int e = 0; e < 8; ++e) o[nf][e] *= alv[e];
    }
    // O += P (16xBc) @ V (Bc x Dh)
#pragma unroll
    for (int ks = 0; ks < Bc; ks += 32) {
      v16bf ap = load_frag_a(sP[wave] + ks, LP);
#pragma unroll
      for (int nf = 0; nf < 4; ++nf) {
        v16bf bv = load_frag_bT(sVt + (nf * 16) * LP + ks, LP);
        o[nf] = wmma_bf16(ap, bv, o[nf]);
      }
    }
    __syncthreads();
  }

  // normalize and write out (row-major [tok][H*Dh])
  {
    const float4* lp = (const float4*)&sL[wave][half * 8];
    float4 l0 = lp[0], l1 = lp[1];
    float lv[8] = {l0.x, l0.y, l0.z, l0.w, l1.x, l1.y, l1.z, l1.w};
#pragma unroll
    for (int nf = 0; nf < 4; ++nf)
#pragma unroll
      for (int e = 0; e < 8; ++e) {
        int r = e + (half << 3);
        float v = o[nf][e] / lv[e];
        O[base + (size_t)(qt * Br + wave * 16 + r) * rs + nf * 16 + (lane & 15)] =
            f2bf(v);
      }
  }
}

// ---------- host launcher ----------
extern "C" void kernel_launch(void* const* d_in, const int* in_sizes, int n_in,
                              void* d_out, int out_size, void* d_ws, size_t ws_size,
                              hipStream_t stream) {
  (void)in_sizes; (void)n_in; (void)out_size; (void)ws_size;
  const float* x  = (const float*)d_in[0];
  const float* Wq = (const float*)d_in[1];
  const float* bq = (const float*)d_in[2];
  const float* Wl = (const float*)d_in[3];
  const float* bl = (const float*)d_in[4];
  const float* Wk = (const float*)d_in[5];
  const float* bk = (const float*)d_in[6];
  const float* Wv = (const float*)d_in[7];
  const float* bv = (const float*)d_in[8];
  const float* Wo = (const float*)d_in[9];
  const float* bo = (const float*)d_in[10];
  float* out = (float*)d_out;

  const int NT = NTOK, D = CFG_D, L = CFG_L;
  char* ws = (char*)d_ws;
  size_t off = 0;
  auto alloc = [&](size_t elems) {
    bf16* p = (bf16*)(ws + off);
    off += elems * sizeof(bf16);
    return p;
  };
  bf16* xb   = alloc((size_t)NT * D);
  bf16* WqT  = alloc((size_t)D * D);   // [N][K] transposed weights
  bf16* WlT  = alloc((size_t)D * L);
  bf16* WkT  = alloc((size_t)L * D);
  bf16* WvT  = alloc((size_t)L * D);
  bf16* WoT  = alloc((size_t)D * D);
  bf16* Qb   = alloc((size_t)NT * D);
  bf16* Lb   = alloc((size_t)NT * L);
  bf16* Kb   = alloc((size_t)NT * D);
  bf16* Vtb  = alloc((size_t)NT * D);  // [B][H][Dh][S]
  bf16* Ob   = alloc((size_t)NT * D);

  f32_to_bf16_kernel<<<(NT * D + 255) / 256, 256, 0, stream>>>(x, xb, NT * D);
  auto cvtT = [&](const float* src, bf16* dst, int R, int C) {
    f32_to_bf16_T_kernel<<<(R * C + 255) / 256, 256, 0, stream>>>(src, dst, R, C);
  };
  cvtT(Wq, WqT, D, D);
  cvtT(Wl, WlT, D, L);
  cvtT(Wk, WkT, L, D);
  cvtT(Wv, WvT, L, D);
  cvtT(Wo, WoT, D, D);

  // Q projection
  gemm_bf16_kernel<0><<<dim3((NT / 128) * (D / 64)), 128, 0, stream>>>(
      xb, WqT, bq, Qb, nullptr, NT, D, D);
  // compressed KV latent
  gemm_bf16_kernel<0><<<dim3((NT / 128) * (L / 64)), 128, 0, stream>>>(
      xb, WlT, bl, Lb, nullptr, NT, L, D);
  // latent -> K
  gemm_bf16_kernel<0><<<dim3((NT / 128) * (D / 64)), 128, 0, stream>>>(
      Lb, WkT, bk, Kb, nullptr, NT, D, L);
  // latent -> V (stored transposed for flash)
  gemm_bf16_kernel<2><<<dim3((NT / 128) * (D / 64)), 128, 0, stream>>>(
      Lb, WvT, bv, Vtb, nullptr, NT, D, L);

  mla_flash_kernel<<<CFG_B * CFG_H * (CFG_S / 64), 128, 0, stream>>>(Qb, Kb, Vtb, Ob);

  // output projection (f32)
  gemm_bf16_kernel<1><<<dim3((NT / 128) * (D / 64)), 128, 0, stream>>>(
      Ob, WoT, bo, nullptr, out, NT, D, D);
}
